// SimpleGRU_65609920413979
// MI455X (gfx1250) — matrix operand order repaired and checked by reading, not compiled, not run on hardware
//
#include <hip/hip_runtime.h>
#include <math.h>

#define B_   4096
#define S_   64
#define H_   256
#define G3_  768   // 3*H

typedef __attribute__((ext_vector_type(16))) __bf16 v16bf;
typedef __attribute__((ext_vector_type(8)))  __bf16 v8bf;
typedef __attribute__((ext_vector_type(8)))  float  v8f;

__device__ __forceinline__ float sigmoidf_(float x) {
  return 1.0f / (1.0f + __expf(-x));
}

__device__ __forceinline__ __bf16 to_bf16(float f) {
  union { float f; unsigned int u; } v; v.f = f;
  unsigned int r = 0x7fffu + ((v.u >> 16) & 1u);   // round-to-nearest-even
  unsigned short h = (unsigned short)((v.u + r) >> 16);
  union { unsigned short s; __bf16 b; } o; o.s = h;
  return o.b;
}

// One K-step worth of register fragments for a 16(M)x96(N) wave tile.
struct Frags {
  v16bf a;      // A: 16x32 bf16 fragment (this wave's 16 rows)
  v16bf b[6];   // B: six 32x16 bf16 fragments (96 N-columns)
};

__device__ __forceinline__ void load_frags(const __bf16* __restrict__ arow,
                                           const __bf16* __restrict__ wbase,
                                           int k0, int g, Frags& f) {
  // A fragment: halves 0-7 <- K=k0+16g.., halves 8-15 <- K=k0+16g+8..
  v8bf alo = *(const v8bf*)(arow + k0 + g * 16);
  v8bf ahi = *(const v8bf*)(arow + k0 + g * 16 + 8);
  #pragma unroll
  for (int i = 0; i < 8; i++) { f.a[i] = alo[i]; f.a[8 + i] = ahi[i]; }
  // B fragments: 16 contiguous K values starting at k0 + 16*g (32B load each)
  #pragma unroll
  for (int tn = 0; tn < 6; tn++)
    f.b[tn] = *(const v16bf*)(wbase + (size_t)tn * 16 * H_ + k0 + g * 16);
}

// Software-pipelined 16x96 GEMM tile: acc[tn] += h[m0:m0+16,:] @ w^T[:,n0+16tn:..]
// Fragment loads for K-step kk+1 are issued before the WMMAs of step kk, so
// each WMMA group waits only on loads issued a full iteration earlier.
__device__ __forceinline__ void gemm_tile(const __bf16* __restrict__ h_b,
                                          const __bf16* __restrict__ wbf,
                                          int m0, int n0, int lane,
                                          v8f acc[6]) {
  const int g   = lane >> 4;
  const int l15 = lane & 15;
  const __bf16* arow  = h_b + (size_t)(m0 + l15) * H_;
  const __bf16* wbase = wbf + (size_t)(n0 + l15) * H_;

  Frags f[2];
  load_frags(arow, wbase, 0, g, f[0]);
  #pragma unroll
  for (int kk = 0; kk < 8; kk++) {               // K = 256 in 8 steps of 32
    if (kk < 7) load_frags(arow, wbase, (kk + 1) * 32, g, f[(kk + 1) & 1]);
    Frags& cur = f[kk & 1];
    #pragma unroll
    for (int tn = 0; tn < 6; tn++)
      acc[tn] = __builtin_amdgcn_wmma_f32_16x16x32_bf16(
          false, cur.a, false, cur.b[tn], (short)0, acc[tn], false, false);
  }
}

// ---- setup: truncate w_hh to bf16 once ----
__global__ void k_prep_w(const float* __restrict__ w_hh, __bf16* __restrict__ wbf) {
  int i = blockIdx.x * blockDim.x + threadIdx.x;
  if (i < G3_ * H_) wbf[i] = to_bf16(w_hh[i]);
}

__global__ void k_zero_h(float* __restrict__ hf, __bf16* __restrict__ hb) {
  int i = blockIdx.x * blockDim.x + threadIdx.x;
  if (i < B_ * H_) { hf[i] = 0.0f; hb[i] = to_bf16(0.0f); }
}

// ---- fused encoder GRU step: gh = h@w_hh^T via bf16 WMMA, then gates ----
// block = 256 threads (8 wave32); block owns rows [16*blockIdx.x, +16), all 768 cols.
__global__ __launch_bounds__(256)
void k_enc_step(const float*  __restrict__ h_in_f,
                const __bf16* __restrict__ h_in_b,
                float*        __restrict__ h_out_f,
                __bf16*       __restrict__ h_out_b,
                const __bf16* __restrict__ wbf,
                const float*  __restrict__ x,
                const float*  __restrict__ w_ih,
                const float*  __restrict__ b_ih,
                const float*  __restrict__ b_hh,
                int t)
{
  __shared__ float lds_gh[16 * 772];           // stride 772 avoids bank aliasing
  const int tid  = threadIdx.x;
  const int wave = tid >> 5;
  const int lane = tid & 31;
  const int g    = lane >> 4;
  const int l15  = lane & 15;
  const int m0   = blockIdx.x * 16;
  const int n0   = wave * 96;                  // each wave: 6 N-tiles of 16

  v8f acc[6];
  #pragma unroll
  for (int i = 0; i < 6; i++)
    #pragma unroll
    for (int r = 0; r < 8; r++) acc[i][r] = 0.0f;

  gemm_tile(h_in_b, wbf, m0, n0, lane, acc);

  // spill gh tile to LDS: C/D layout -> row = r + 8*g, col = n
  #pragma unroll
  for (int tn = 0; tn < 6; tn++) {
    const int n = n0 + tn * 16 + l15;
    #pragma unroll
    for (int r = 0; r < 8; r++)
      lds_gh[(r + 8 * g) * 772 + n] = acc[tn][r];
  }
  __syncthreads();

  // fused gate math (f32): thread 'hc' handles all 16 rows of its hidden column
  const int hc = tid;
  const float wir = w_ih[hc], wiz = w_ih[H_ + hc], win = w_ih[2 * H_ + hc];
  const float bir = b_ih[hc], biz = b_ih[H_ + hc], bin_ = b_ih[2 * H_ + hc];
  const float bhr = b_hh[hc], bhz = b_hh[H_ + hc], bhn = b_hh[2 * H_ + hc];
  #pragma unroll 4
  for (int row = 0; row < 16; row++) {
    const int b = m0 + row;
    const float xv = x[(size_t)b * S_ + t];          // x: [B, S, 1]
    const float ghr = lds_gh[row * 772 + hc]          + bhr;
    const float ghz = lds_gh[row * 772 + H_ + hc]     + bhz;
    const float ghn = lds_gh[row * 772 + 2 * H_ + hc] + bhn;
    const float hprev = h_in_f[(size_t)b * H_ + hc];
    const float r = sigmoidf_(xv * wir + bir + ghr);
    const float z = sigmoidf_(xv * wiz + biz + ghz);
    const float n = tanhf(xv * win + bin_ + r * ghn);
    const float hn = (1.0f - z) * n + z * hprev;
    h_out_f[(size_t)b * H_ + hc] = hn;
    h_out_b[(size_t)b * H_ + hc] = to_bf16(hn);
  }
}

// ---- decoder constant: gh_fixed = hidden @ w_hh^T + b_hh (one GEMM) ----
__global__ __launch_bounds__(256)
void k_gh_fixed(const __bf16* __restrict__ h_b,
                const __bf16* __restrict__ wbf,
                const float*  __restrict__ b_hh,
                float*        __restrict__ ghF)
{
  const int tid  = threadIdx.x;
  const int wave = tid >> 5;
  const int lane = tid & 31;
  const int g    = lane >> 4;
  const int l15  = lane & 15;
  const int m0   = blockIdx.x * 16;
  const int n0   = wave * 96;

  v8f acc[6];
  #pragma unroll
  for (int i = 0; i < 6; i++)
    #pragma unroll
    for (int r = 0; r < 8; r++) acc[i][r] = 0.0f;

  gemm_tile(h_b, wbf, m0, n0, lane, acc);

  #pragma unroll
  for (int tn = 0; tn < 6; tn++) {
    const int n = n0 + tn * 16 + l15;
    const float bh = b_hh[n];
    #pragma unroll
    for (int r = 0; r < 8; r++)
      ghF[(size_t)(m0 + r + 8 * g) * G3_ + n] = acc[tn][r] + bh;
  }
}

// ---- decoder head: out[b] = h_t[b,:]·w_out + b_out ; one wave per row ----
__global__ __launch_bounds__(256)
void k_head(const float* __restrict__ h, const float* __restrict__ w_out,
            const float* __restrict__ b_out, float* __restrict__ out_t,
            float* __restrict__ d_out, int t, int T)
{
  const int wave = threadIdx.x >> 5;
  const int lane = threadIdx.x & 31;
  const int row  = blockIdx.x * 8 + wave;
  float s = 0.0f;
  #pragma unroll
  for (int i = 0; i < 8; i++) {
    const int k = lane + i * 32;
    s += h[(size_t)row * H_ + k] * w_out[k];
  }
  #pragma unroll
  for (int m = 16; m >= 1; m >>= 1) s += __shfl_xor(s, m, 32);
  if (lane == 0) {
    const float o = s + b_out[0];
    out_t[row] = o;
    d_out[(size_t)row * T + t] = o;                  // out layout [B, T]
  }
}

// ---- decoder cell: pure elementwise (gh is the precomputed constant) ----
__global__ __launch_bounds__(256)
void k_dec_cell(const float* __restrict__ out_t,
                const float* __restrict__ ghF,
                const float* __restrict__ hidden,
                const float* __restrict__ w_ih,
                const float* __restrict__ b_ih,
                float* __restrict__ h_out)
{
  const int b  = blockIdx.x;
  const int hc = threadIdx.x;
  const float o   = out_t[b];
  const float ghr = ghF[(size_t)b * G3_ + hc];
  const float ghz = ghF[(size_t)b * G3_ + H_ + hc];
  const float ghn = ghF[(size_t)b * G3_ + 2 * H_ + hc];
  const float r = sigmoidf_(o * w_ih[hc]          + b_ih[hc]          + ghr);
  const float z = sigmoidf_(o * w_ih[H_ + hc]     + b_ih[H_ + hc]     + ghz);
  const float n = tanhf   (o * w_ih[2 * H_ + hc]  + b_ih[2 * H_ + hc] + r * ghn);
  h_out[(size_t)b * H_ + hc] = (1.0f - z) * n + z * hidden[(size_t)b * H_ + hc];
}

extern "C" void kernel_launch(void* const* d_in, const int* in_sizes, int n_in,
                              void* d_out, int out_size, void* d_ws, size_t ws_size,
                              hipStream_t stream) {
  (void)in_sizes; (void)n_in; (void)ws_size;
  const float* x     = (const float*)d_in[0];
  const float* w_ih  = (const float*)d_in[1];
  const float* w_hh  = (const float*)d_in[2];
  const float* b_ih  = (const float*)d_in[3];
  const float* b_hh  = (const float*)d_in[4];
  const float* w_out = (const float*)d_in[5];
  const float* b_out = (const float*)d_in[6];
  float* out = (float*)d_out;
  const int T = out_size / B_;                       // == task_level (42)

  // workspace carve-up (~32.5 MB total)
  char* ws = (char*)d_ws;
  size_t off = 0;
  auto carve = [&](size_t bytes) -> void* {
    void* p = ws + off;
    off += (bytes + 255) & ~(size_t)255;
    return p;
  };
  __bf16* wbf  = (__bf16*)carve((size_t)G3_ * H_ * 2);
  float*  hAf  = (float*) carve((size_t)B_ * H_ * 4);
  float*  hBf  = (float*) carve((size_t)B_ * H_ * 4);
  __bf16* hAb  = (__bf16*)carve((size_t)B_ * H_ * 2);
  __bf16* hBb  = (__bf16*)carve((size_t)B_ * H_ * 2);
  float*  ghF  = (float*) carve((size_t)B_ * G3_ * 4);
  float*  dA   = (float*) carve((size_t)B_ * H_ * 4);
  float*  dB   = (float*) carve((size_t)B_ * H_ * 4);
  float*  outt = (float*) carve((size_t)B_ * 4);

  k_prep_w<<<(G3_ * H_ + 255) / 256, 256, 0, stream>>>(w_hh, wbf);
  k_zero_h<<<(B_ * H_ + 255) / 256, 256, 0, stream>>>(hAf, hAb);

  // ---- encoder scan: 64 sequential WMMA-GEMM + fused-gate steps ----
  const float*  hf_in = hAf;  const __bf16* hb_in = hAb;
  float*        hf_out = hBf; __bf16*       hb_out = hBb;
  for (int t = 0; t < S_; t++) {
    k_enc_step<<<B_ / 16, 256, 0, stream>>>(hf_in, hb_in, hf_out, hb_out,
                                            wbf, x, w_ih, b_ih, b_hh, t);
    const float* tf = hf_in;  hf_in = hf_out;  hf_out = (float*)tf;
    const __bf16* tb = hb_in; hb_in = hb_out;  hb_out = (__bf16*)tb;
  }
  // final hidden now in (hf_in, hb_in)

  // ---- decoder: one GEMM for the step-invariant gh, then scalar loop ----
  k_gh_fixed<<<B_ / 16, 256, 0, stream>>>(hb_in, wbf, b_hh, ghF);

  const float* hcur = hf_in;                         // h_0 = hidden
  float* dbuf[2] = { dA, dB };
  for (int t = 0; t < T; t++) {
    k_head<<<B_ / 8, 256, 0, stream>>>(hcur, w_out, b_out, outt, out, t, T);
    if (t + 1 < T) {
      float* hn = dbuf[t & 1];
      k_dec_cell<<<B_, 256, 0, stream>>>(outt, ghF, hf_in, w_ih, b_ih, hn);
      hcur = hn;
    }
  }
}
